// MoEWrapper_79199196938470
// MI455X (gfx1250) — matrix operand order repaired and checked by reading, not compiled, not run on hardware
//
#include <hip/hip_runtime.h>
#include <stdint.h>

// MoE: T=16384 tokens, H=768, F=3072, E=8 experts, top-2 routing.
// Pipeline: router -> prefix offsets -> grouped FC1 (gather, bf16 WMMA, GELU,
// gate-weight fold) -> grouped FC2 (TDM async A-panel, bf16 WMMA, scatter-add).
// Workspace layout (needs ~203MB):
//   [0,32)    : per-expert counts (zeroed each launch)
//   [32,64)   : per-expert exclusive prefix offsets
//   [64, +512K): token lists  (E*T int)
//   next 512K : gate weights  (E*T float)
//   next 201MB: h1 intermediate, bf16, [2T rows x F]

#define T_TOK 16384
#define H_DIM 768
#define F_DIM 3072
#define N_EXP 8

typedef __attribute__((ext_vector_type(16))) __bf16    bf16x16;
typedef __attribute__((ext_vector_type(8)))  float     f32x8;
typedef __attribute__((ext_vector_type(4)))  float     f32x4;
typedef __attribute__((ext_vector_type(4)))  uint32_t  u32x4;
typedef __attribute__((ext_vector_type(8)))  int       i32x8;
typedef __attribute__((ext_vector_type(4)))  int       i32x4;

union Frag { u32x4 q[2]; bf16x16 v; };

#define LDS_STRIDE 40  // bf16 elems per LDS row: 32 data + 8 pad (80B, 16B aligned)

__device__ __forceinline__ uint16_t f2bf(float f) {
  uint32_t u = __float_as_uint(f);
  u = (u + 0x7FFFu + ((u >> 16) & 1u)) >> 16;  // round-to-nearest-even
  return (uint16_t)u;
}
__device__ __forceinline__ uint32_t pack_bf2(float a, float b) {
  return (uint32_t)f2bf(a) | ((uint32_t)f2bf(b) << 16);
}
__device__ __forceinline__ float gelu_tanh(float v) {
  float u = 0.7978845608028654f * (v + 0.044715f * v * v * v);
  return 0.5f * v * (1.0f + tanhf(u));
}

// ---------------------------------------------------------------- router ----
__global__ void __launch_bounds__(256) moe_router(
    const float* __restrict__ x, const float* __restrict__ Wr,
    const float* __restrict__ br, int* __restrict__ cnt,
    int* __restrict__ list, float* __restrict__ wl) {
  __shared__ float sWr[H_DIM * N_EXP];
  __shared__ float sbr[N_EXP];
  int tid = threadIdx.x;
  for (int i = tid; i < H_DIM * N_EXP; i += 256) sWr[i] = Wr[i];
  if (tid < N_EXP) sbr[tid] = br[tid];
  __syncthreads();

  int lane = tid & 31;
  int wave = (blockIdx.x * 256 + tid) >> 5;
  int nwaves = (gridDim.x * 256) >> 5;

  for (int t = wave; t < T_TOK; t += nwaves) {
    float acc[N_EXP];
#pragma unroll
    for (int e = 0; e < N_EXP; ++e) acc[e] = 0.f;
    const float* xr = x + (size_t)t * H_DIM;
#pragma unroll 4
    for (int kk = 0; kk < H_DIM / 32; ++kk) {
      int k = kk * 32 + lane;
      float xv = xr[k];
#pragma unroll
      for (int e = 0; e < N_EXP; ++e) acc[e] += xv * sWr[k * N_EXP + e];
    }
#pragma unroll
    for (int e = 0; e < N_EXP; ++e) {
#pragma unroll
      for (int off = 16; off > 0; off >>= 1)
        acc[e] += __shfl_xor(acc[e], off, 32);
      acc[e] += sbr[e];
    }
    // softmax over 8 logits (all lanes hold full vector)
    float mx = acc[0];
#pragma unroll
    for (int e = 1; e < N_EXP; ++e) mx = fmaxf(mx, acc[e]);
    float p[N_EXP], sum = 0.f;
#pragma unroll
    for (int e = 0; e < N_EXP; ++e) { p[e] = __expf(acc[e] - mx); sum += p[e]; }
    float inv = 1.f / sum;
#pragma unroll
    for (int e = 0; e < N_EXP; ++e) p[e] *= inv;
    // top-2, lower index wins ties (matches lax.top_k)
    int i1 = 0; float p1 = p[0];
#pragma unroll
    for (int e = 1; e < N_EXP; ++e) if (p[e] > p1) { p1 = p[e]; i1 = e; }
    int i2 = (i1 == 0) ? 1 : 0; float p2 = p[i2];
#pragma unroll
    for (int e = 0; e < N_EXP; ++e)
      if (e != i1 && p[e] > p2) { p2 = p[e]; i2 = e; }
    float wn = 1.f / (p1 + p2);
    if (lane == 0) {
      int pos1 = atomicAdd(&cnt[i1], 1);
      list[i1 * T_TOK + pos1] = t; wl[i1 * T_TOK + pos1] = p1 * wn;
      int pos2 = atomicAdd(&cnt[i2], 1);
      list[i2 * T_TOK + pos2] = t; wl[i2 * T_TOK + pos2] = p2 * wn;
    }
  }
}

// ----------------------------------------------------------- prefix scan ----
__global__ void moe_offsets(const int* __restrict__ cnt, int* __restrict__ offs) {
  if (threadIdx.x == 0) {
    int run = 0;
#pragma unroll
    for (int e = 0; e < N_EXP; ++e) { offs[e] = run; run += cnt[e]; }
  }
}

// ------------------------------------------------------------------- FC1 ----
// h1[row, fc+n] = gelu(x_gathered @ Wfc[e] + bfc[e]) * gate_w,  bf16 out.
__global__ void __launch_bounds__(256) moe_fc1(
    const float* __restrict__ x, const float* __restrict__ Wfc,
    const float* __restrict__ bfc, const int* __restrict__ cnt,
    const int* __restrict__ offs, const int* __restrict__ list,
    const float* __restrict__ wl, uint16_t* __restrict__ h1) {
  int e = blockIdx.z;
  int cnt_e = cnt[e];
  int m0 = blockIdx.x * 128;
  if (m0 >= cnt_e) return;  // block-uniform early exit (EXEC stays all-ones)
  int fc = blockIdx.y * 128;
  int hbase = offs[e];

  __shared__ alignas(16) uint16_t sA[128 * LDS_STRIDE];
  __shared__ alignas(16) uint16_t sB[128 * LDS_STRIDE];
  __shared__ int   sT[128];
  __shared__ float sW[128];

  int tid = threadIdx.x;
  if (tid < 128) {
    int r = m0 + tid;
    bool v = r < cnt_e;
    sT[tid] = v ? list[e * T_TOK + r] : 0;
    sW[tid] = v ? wl[e * T_TOK + r] : 0.f;
  }
  __syncthreads();

  int lane = tid & 31, wv = tid >> 5, hf = lane >> 4, l15 = lane & 15;
  int am = tid >> 1, akh = (tid & 1) * 16;          // A fill: row, k-half
  int bn = tid & 127, bkb = (tid >> 7) * 16;        // B fill: col, k-base
  const float* Wb = Wfc + (size_t)e * H_DIM * F_DIM + fc + bn;

  f32x8 zero = {0.f, 0.f, 0.f, 0.f, 0.f, 0.f, 0.f, 0.f};
  f32x8 acc[8];
#pragma unroll
  for (int nt = 0; nt < 8; ++nt) acc[nt] = zero;

  for (int k0 = 0; k0 < H_DIM; k0 += 32) {
    { // A panel: gather x rows, f32 -> bf16
      const float* p = x + (size_t)sT[am] * H_DIM + k0 + akh;
      f32x4 f0 = *(const f32x4*)(p + 0);
      f32x4 f1 = *(const f32x4*)(p + 4);
      f32x4 f2 = *(const f32x4*)(p + 8);
      f32x4 f3 = *(const f32x4*)(p + 12);
      u32x4 q0 = {pack_bf2(f0[0], f0[1]), pack_bf2(f0[2], f0[3]),
                  pack_bf2(f1[0], f1[1]), pack_bf2(f1[2], f1[3])};
      u32x4 q1 = {pack_bf2(f2[0], f2[1]), pack_bf2(f2[2], f2[3]),
                  pack_bf2(f3[0], f3[1]), pack_bf2(f3[2], f3[3])};
      uint16_t* d = &sA[am * LDS_STRIDE + akh];
      *(u32x4*)d = q0;
      *(u32x4*)(d + 8) = q1;
    }
    if (k0 + 32 < H_DIM)
      __builtin_prefetch(&Wb[(size_t)(k0 + 32 + bkb) * F_DIM], 0, 3);
    // B panel stored N-major (transposed): sB[n][k] = Wfc[k0+k][fc+n]
#pragma unroll 4
    for (int kk = 0; kk < 16; ++kk) {
      int k = bkb + kk;
      sB[bn * LDS_STRIDE + k] = f2bf(Wb[(size_t)(k0 + k) * F_DIM]);
    }
    __syncthreads();

    Frag fa;  // 16x32 bf16 A frag: lane holds row M=l15, K = 8h.. and 16+8h..
    const uint16_t* pa = &sA[(wv * 16 + l15) * LDS_STRIDE + 8 * hf];
    fa.q[0] = *(const u32x4*)pa;
    fa.q[1] = *(const u32x4*)(pa + 16);
#pragma unroll
    for (int nt = 0; nt < 8; ++nt) {
      Frag fb;  // 32x16 bf16 B frag: lane holds col N=l15, K = 8h.. and 16+8h..
      const uint16_t* pb = &sB[(nt * 16 + l15) * LDS_STRIDE + 8 * hf];
      fb.q[0] = *(const u32x4*)pb;
      fb.q[1] = *(const u32x4*)(pb + 16);
      acc[nt] = __builtin_amdgcn_wmma_f32_16x16x32_bf16(
          false, fa.v, false, fb.v, (short)0, acc[nt], false, false);
    }
    __syncthreads();
  }

  // epilogue: +bfc, GELU, fold gate weight, store bf16
#pragma unroll
  for (int nt = 0; nt < 8; ++nt) {
    int n = nt * 16 + l15;
    float bias = bfc[(size_t)e * F_DIM + fc + n];
#pragma unroll
    for (int r = 0; r < 8; ++r) {
      int ml = wv * 16 + r + 8 * hf;  // C layout: VGPR r -> M = r + 8*half
      if (m0 + ml < cnt_e) {
        float v = gelu_tanh(acc[nt][r] + bias) * sW[ml];
        h1[(size_t)(hbase + m0 + ml) * F_DIM + fc + n] = f2bf(v);
      }
    }
  }
}

// ------------------------------------------------------------------- FC2 ----
// out[token, hc+n] += h1 @ Wproj[e] + gate_w * bproj[e]   (scatter-add)
// A-panel (contiguous bf16 rows in packed h1 space) is DMA'd by the Tensor
// Data Mover: 128x32-elem tile, LDS pad 16B per 64B row -> pitch 80B.
__global__ void __launch_bounds__(256) moe_fc2(
    const uint16_t* __restrict__ h1, const float* __restrict__ Wproj,
    const float* __restrict__ bproj, const int* __restrict__ cnt,
    const int* __restrict__ offs, const int* __restrict__ list,
    const float* __restrict__ wl, float* __restrict__ out) {
  int e = blockIdx.z;
  int cnt_e = cnt[e];
  int m0 = blockIdx.x * 128;
  if (m0 >= cnt_e) return;
  int hc = blockIdx.y * 128;
  int hbase = offs[e];

  __shared__ alignas(16) uint16_t sA[128 * LDS_STRIDE];
  __shared__ alignas(16) uint16_t sB[128 * LDS_STRIDE];
  __shared__ int   sT[128];
  __shared__ float sW[128];

  int tid = threadIdx.x;
  if (tid < 128) {
    int r = m0 + tid;
    bool v = r < cnt_e;
    sT[tid] = v ? list[e * T_TOK + r] : 0;
    sW[tid] = v ? wl[e * T_TOK + r] : 0.f;
  }
  __syncthreads();

  int lane = tid & 31, wv = tid >> 5, hf = lane >> 4, l15 = lane & 15;
  int bn = tid & 127, bkb = (tid >> 7) * 16;
  const float* Wb = Wproj + (size_t)e * F_DIM * H_DIM + hc + bn;

  uint32_t lds_a = (uint32_t)(uintptr_t)&sA[0];

  f32x8 zero = {0.f, 0.f, 0.f, 0.f, 0.f, 0.f, 0.f, 0.f};
  f32x8 acc[8];
#pragma unroll
  for (int nt = 0; nt < 8; ++nt) acc[nt] = zero;

  for (int k0 = 0; k0 < F_DIM; k0 += 32) {
    // --- TDM: async-load A tile [128 rows x 32 bf16] into sA (wave 0 only;
    //     TDM ignores EXEC, one issue per workgroup). tensor_dim1 = 2T rows
    //     gives hardware zero-fill for the tail tile past the h1 buffer.
    if (wv == 0) {
      uint64_t ga = (uint64_t)(uintptr_t)(h1 + (size_t)(hbase + m0) * F_DIM + k0);
      u32x4 g0;
      g0[0] = 1u;                                   // count=1, user mode
      g0[1] = lds_a;                                // lds_addr (bytes)
      g0[2] = (uint32_t)ga;                         // global_addr[31:0]
      g0[3] = (uint32_t)((ga >> 32) & 0x1FFFFFFu)   // global_addr[56:32]
              | (2u << 30);                         // type=2 (image)
      i32x8 g1;
      g1[0] = (int)((1u << 16)      // data_size=1 -> 2 bytes (bf16)
                  | (1u << 20)      // pad_enable
                  | (3u << 22)      // pad_interval: every 16 DWORDs (64B row)
                  | (3u << 25));    // pad_amount: 4 DWORDs (16B) -> 80B pitch
      g1[1] = (int)(((uint32_t)F_DIM & 0xFFFFu) << 16);       // tensor_dim0 lo
      g1[2] = (int)(((uint32_t)(2 * T_TOK) & 0xFFFFu) << 16); // tensor_dim1 lo
      g1[3] = (int)(32u << 16);                               // tile_dim0 = 32
      g1[4] = 128;                                            // tile_dim1 = 128
      g1[5] = F_DIM;                                          // dim0_stride lo
      g1[6] = 0;
      g1[7] = 0;
      i32x4 gz = {0, 0, 0, 0};          // 2-D tile: groups 2/3 unused
      i32x8 gz8 = {0, 0, 0, 0, 0, 0, 0, 0};
      __builtin_amdgcn_tensor_load_to_lds(g0, g1, gz, gz, gz8, 0);
    }
    if (k0 + 32 < F_DIM)
      __builtin_prefetch(&Wb[(size_t)(k0 + 32 + bkb) * H_DIM], 0, 3);
    // --- overlap: B panel f32->bf16 convert, N-major: sB[n][k] = Wproj[k0+k][hc+n]
#pragma unroll 4
    for (int kk = 0; kk < 16; ++kk) {
      int k = bkb + kk;
      sB[bn * LDS_STRIDE + k] = f2bf(Wb[(size_t)(k0 + k) * H_DIM]);
    }
    if (wv == 0) __builtin_amdgcn_s_wait_tensorcnt(0);  // issuing wave drains TDM
    __syncthreads();

    Frag fa;
    const uint16_t* pa = &sA[(wv * 16 + l15) * LDS_STRIDE + 8 * hf];
    fa.q[0] = *(const u32x4*)pa;
    fa.q[1] = *(const u32x4*)(pa + 16);
#pragma unroll
    for (int nt = 0; nt < 8; ++nt) {
      Frag fb;
      const uint16_t* pb = &sB[(nt * 16 + l15) * LDS_STRIDE + 8 * hf];
      fb.q[0] = *(const u32x4*)pb;
      fb.q[1] = *(const u32x4*)(pb + 16);
      acc[nt] = __builtin_amdgcn_wmma_f32_16x16x32_bf16(
          false, fa.v, false, fb.v, (short)0, acc[nt], false, false);
    }
    __syncthreads();
  }

  // epilogue: + w*bproj, scatter-add (exactly 2 adds/element -> deterministic)
#pragma unroll
  for (int nt = 0; nt < 8; ++nt) {
    int n = nt * 16 + l15;
    int col = hc + n;
    float bias = bproj[(size_t)e * H_DIM + col];
#pragma unroll
    for (int r = 0; r < 8; ++r) {
      int ml = wv * 16 + r + 8 * hf;
      if (m0 + ml < cnt_e) {
        float v = acc[nt][r] + sW[ml] * bias;
        atomicAdd(&out[(size_t)sT[ml] * H_DIM + col], v);
      }
    }
  }
}

// ---------------------------------------------------------------- launch ----
extern "C" void kernel_launch(void* const* d_in, const int* in_sizes, int n_in,
                              void* d_out, int out_size, void* d_ws, size_t ws_size,
                              hipStream_t stream) {
  const float* x     = (const float*)d_in[0];
  const float* Wr    = (const float*)d_in[1];
  const float* br    = (const float*)d_in[2];
  const float* Wfc   = (const float*)d_in[3];
  const float* bfc   = (const float*)d_in[4];
  const float* Wproj = (const float*)d_in[5];
  const float* bproj = (const float*)d_in[6];
  float* out = (float*)d_out;

  char* wsb = (char*)d_ws;
  int*      cnt  = (int*)(wsb + 0);
  int*      offs = (int*)(wsb + 32);
  int*      list = (int*)(wsb + 64);
  float*    wl   = (float*)(wsb + 64 + (size_t)N_EXP * T_TOK * 4);
  uint16_t* h1   = (uint16_t*)(wsb + 64 + (size_t)2 * N_EXP * T_TOK * 4);

  (void)hipMemsetAsync(cnt, 0, 32, stream);
  (void)hipMemsetAsync(out, 0, (size_t)T_TOK * H_DIM * sizeof(float), stream);

  moe_router<<<64, 256, 0, stream>>>(x, Wr, br, cnt, list, wl);
  moe_offsets<<<1, 32, 0, stream>>>(cnt, offs);
  moe_fc1<<<dim3(T_TOK / 128, F_DIM / 128, N_EXP), 256, 0, stream>>>(
      x, Wfc, bfc, cnt, offs, list, wl, h1);
  moe_fc2<<<dim3(T_TOK / 128, H_DIM / 128, N_EXP), 256, 0, stream>>>(
      h1, Wproj, bproj, cnt, offs, list, wl, out);
}
